// DebertaBare_39926015984257
// MI455X (gfx1250) — matrix operand order repaired and checked by reading, not compiled, not run on hardware
//
#include <hip/hip_runtime.h>
#include <hip/hip_bf16.h>
#include <math.h>

typedef __bf16 bf16_t;
typedef __attribute__((ext_vector_type(16))) __bf16 v16bf;
typedef __attribute__((ext_vector_type(8)))  float  v8f;

#define WG 256
#define BM 128
#define BN 64
#define BK 64
#define LDSA_STRIDE 72   // 144B rows: keeps every 8-elem chunk 16B aligned
#define LDSB_STRIDE 72

// ---------------------------------------------------------------------------
// CDNA5 async global->LDS helpers (no VGPR payload; tracked by ASYNCcnt)
// ---------------------------------------------------------------------------
__device__ __forceinline__ unsigned lds_off32(const void* p) {
    // LDS aperture: addr[31:0] is the byte offset within workgroup LDS
    return (unsigned)(unsigned long long)p;
}
__device__ __forceinline__ void async_load_b128(unsigned lds_off, const void* gptr) {
    asm volatile("global_load_async_to_lds_b128 %0, %1, off"
                 :: "v"(lds_off), "v"(gptr) : "memory");
}
__device__ __forceinline__ void wait_async0() {
    asm volatile("s_wait_asynccnt 0x0" ::: "memory");
}

// ---------------------------------------------------------------------------
// Transpose + fp32 -> bf16  (weights -> [N,K] bf16 so all GEMMs are NT form)
// ---------------------------------------------------------------------------
__global__ void k_transpose_cvt(const float* __restrict__ in, bf16_t* __restrict__ out,
                                int R, int C) {
    __shared__ float tile[32][33];
    int c0 = blockIdx.x * 32, r0 = blockIdx.y * 32;
    int tx = threadIdx.x & 31, ty = threadIdx.x >> 5;   // 32 x 8
    for (int i = 0; i < 32; i += 8) {
        int r = r0 + ty + i, c = c0 + tx;
        tile[ty + i][tx] = (r < R && c < C) ? in[(long long)r * C + c] : 0.f;
    }
    __syncthreads();
    for (int i = 0; i < 32; i += 8) {
        int c = c0 + ty + i, r = r0 + tx;               // out[c][r] = in[r][c]
        if (c < C && r < R) out[(long long)c * R + r] = (bf16_t)tile[tx][ty + i];
    }
}

// ---------------------------------------------------------------------------
// Plain fp32 -> bf16 (rel_emb)
// ---------------------------------------------------------------------------
__global__ void k_cvt_bf16(const float* __restrict__ in, bf16_t* __restrict__ out,
                           long long n) {
    long long i = ((long long)blockIdx.x * blockDim.x + threadIdx.x) * 4;
    if (i + 3 < n) {
        float4 v = *(const float4*)(in + i);
        out[i + 0] = (bf16_t)v.x; out[i + 1] = (bf16_t)v.y;
        out[i + 2] = (bf16_t)v.z; out[i + 3] = (bf16_t)v.w;
    } else {
        for (int j = 0; j < 4 && i + j < n; ++j) out[i + j] = (bf16_t)in[i + j];
    }
}

// ---------------------------------------------------------------------------
// Embedding: tok + seg gather, LayerNorm, mask; emit fp32 + bf16 activations
// ---------------------------------------------------------------------------
__global__ void k_embed_ln(const int* __restrict__ ids, const int* __restrict__ segs,
                           const float* __restrict__ mask,
                           const float* __restrict__ tok, const float* __restrict__ seg,
                           const float* __restrict__ g, const float* __restrict__ b,
                           float* __restrict__ h, bf16_t* __restrict__ hb, int Dm) {
    int row = blockIdx.x, t = threadIdx.x;
    int id = ids[row], sg = segs[row];
    float mk = mask[row];
    const float* te = tok + (long long)id * Dm;
    const float* se = seg + (long long)sg * Dm;
    float x[4]; float s = 0.f;
    for (int j = 0; j < 4; ++j) { int d = t + j * WG; x[j] = te[d] + se[d]; s += x[j]; }
    __shared__ float red[WG];
    red[t] = s; __syncthreads();
    for (int o = WG / 2; o > 0; o >>= 1) { if (t < o) red[t] += red[t + o]; __syncthreads(); }
    float mean = red[0] / Dm; __syncthreads();
    float vs = 0.f;
    for (int j = 0; j < 4; ++j) { float d0 = x[j] - mean; vs += d0 * d0; }
    red[t] = vs; __syncthreads();
    for (int o = WG / 2; o > 0; o >>= 1) { if (t < o) red[t] += red[t + o]; __syncthreads(); }
    float rstd = rsqrtf(red[0] / Dm + 1e-12f);
    for (int j = 0; j < 4; ++j) {
        int d = t + j * WG;
        float y = ((x[j] - mean) * rstd * g[d] + b[d]) * mk;
        h [(long long)row * Dm + d] = y;
        hb[(long long)row * Dm + d] = (bf16_t)y;
    }
}

// ---------------------------------------------------------------------------
// Residual + LayerNorm; emit fp32 + bf16
// ---------------------------------------------------------------------------
__global__ void k_ln_residual(const float* __restrict__ x, const float* __restrict__ dlt,
                              const float* __restrict__ g, const float* __restrict__ b,
                              float* __restrict__ yf, bf16_t* __restrict__ yb, int Dm) {
    int row = blockIdx.x, t = threadIdx.x;
    float v[4]; float s = 0.f;
    for (int j = 0; j < 4; ++j) {
        int d = t + j * WG;
        v[j] = x[(long long)row * Dm + d] + dlt[(long long)row * Dm + d];
        s += v[j];
    }
    __shared__ float red[WG];
    red[t] = s; __syncthreads();
    for (int o = WG / 2; o > 0; o >>= 1) { if (t < o) red[t] += red[t + o]; __syncthreads(); }
    float mean = red[0] / Dm; __syncthreads();
    float vs = 0.f;
    for (int j = 0; j < 4; ++j) { float d0 = v[j] - mean; vs += d0 * d0; }
    red[t] = vs; __syncthreads();
    for (int o = WG / 2; o > 0; o >>= 1) { if (t < o) red[t] += red[t + o]; __syncthreads(); }
    float rstd = rsqrtf(red[0] / Dm + 1e-12f);
    for (int j = 0; j < 4; ++j) {
        int d = t + j * WG;
        float y = (v[j] - mean) * rstd * g[d] + b[d];
        yf[(long long)row * Dm + d] = y;
        yb[(long long)row * Dm + d] = (bf16_t)y;
    }
}

// ---------------------------------------------------------------------------
// Fused disentangled-attention softmax:
//   att[q,k] = qk + c2p[q, idx] + p2c[k, idx] + (1-mask[k])*NEG,
//   idx = clip(q-k+512, 0, P-1)  (c2p_idx[q,k] == p2c_idx[k,q])
// ---------------------------------------------------------------------------
__global__ void k_attn_softmax(const float* __restrict__ qk, const float* __restrict__ c2p,
                               const float* __restrict__ p2c, const float* __restrict__ mask,
                               bf16_t* __restrict__ probs, int S, int P) {
    int q = blockIdx.x, hh = blockIdx.y, t = threadIdx.x;
    long long rowq = (long long)hh * S + q;
    float v[4]; float mx = -3.4e38f;
    for (int j = 0; j < 4; ++j) {
        int k = t + j * WG;
        int idx = q - k + 512;
        idx = idx < 0 ? 0 : (idx > P - 1 ? P - 1 : idx);
        float val = qk[rowq * S + k] + c2p[rowq * P + idx]
                  + p2c[((long long)hh * S + k) * P + idx]
                  + (1.0f - mask[k]) * -1e9f;
        v[j] = val; mx = fmaxf(mx, val);
    }
    __shared__ float red[WG];
    red[t] = mx; __syncthreads();
    for (int o = WG / 2; o > 0; o >>= 1) { if (t < o) red[t] = fmaxf(red[t], red[t + o]); __syncthreads(); }
    mx = red[0]; __syncthreads();
    float s = 0.f;
    for (int j = 0; j < 4; ++j) { v[j] = __expf(v[j] - mx); s += v[j]; }
    red[t] = s; __syncthreads();
    for (int o = WG / 2; o > 0; o >>= 1) { if (t < o) red[t] += red[t + o]; __syncthreads(); }
    float inv = 1.0f / red[0];
    for (int j = 0; j < 4; ++j) probs[rowq * S + t + j * WG] = (bf16_t)(v[j] * inv);
}

// ---------------------------------------------------------------------------
// Batched NT bf16 WMMA GEMM:  C[m,n] = alpha * sum_k A[m,k]*B[n,k] + bias[n]
//  - 128x64 C tile / WG (8 waves, wave32); BK=64 = 2 WMMA k-steps per stage
//  - double-buffered LDS fed by GLOBAL_LOAD_ASYNC_TO_LDS_B128 (ASYNCcnt),
//    next stage issued before the 8 WMMAs -> latency hidden, zero staging VGPRs
//  - REQUIRES: M%128==0, N%64==0, K%64==0 (true for every GEMM in this net)
//  - generic (ldcM, ldcN) output strides + fp32/bf16 + optional exact GELU
// ---------------------------------------------------------------------------
struct GemmP {
    const bf16_t* A; const bf16_t* B; void* C; const float* bias;
    int M, N, K;
    long long lda, ldb, ldcM, ldcN;
    long long strA, strB, strC;    // per-blockIdx.z strides (elements)
    float alpha;
    int out_bf16, act_gelu;
};

__launch_bounds__(256)
__global__ void k_gemm_wmma(GemmP p) {
    __shared__ __align__(16) bf16_t sA[2][BM * LDSA_STRIDE];   // 2 x 18 KB
    __shared__ __align__(16) bf16_t sB[2][BN * LDSB_STRIDE];   // 2 x  9 KB

    const int tid  = threadIdx.x;
    const int lane = tid & 31, wave = tid >> 5;
    const int wm = (wave & 3) * 32;      // wave M offset in 128-tile
    const int wn = (wave >> 2) * 32;     // wave N offset in 64-tile
    const long long m0 = (long long)blockIdx.y * BM;
    const long long n0 = (long long)blockIdx.x * BN;
    const bf16_t* A  = p.A + (long long)blockIdx.z * p.strA;
    const bf16_t* Bm = p.B + (long long)blockIdx.z * p.strB;

    // cooperative staging geometry: 8-elem (16B) chunks, 8 chunks per 64-col row
    const int sRow = tid >> 3;           // 0..31
    const int sCol = (tid & 7) * 8;      // 0,8,...,56
    const bf16_t* aG = A  + (m0 + sRow) * p.lda + sCol;   // + 32*i rows, + s*BK cols
    const bf16_t* bG = Bm + (n0 + sRow) * p.ldb + sCol;

    // precomputed per-lane LDS byte offsets for both buffers
    unsigned aL[2], bL[2];
    #pragma unroll
    for (int bfi = 0; bfi < 2; ++bfi) {
        aL[bfi] = lds_off32(&sA[bfi][sRow * LDSA_STRIDE + sCol]);
        bL[bfi] = lds_off32(&sB[bfi][sRow * LDSB_STRIDE + sCol]);
    }

    v8f acc[2][2];
    #pragma unroll
    for (int i = 0; i < 2; ++i)
        #pragma unroll
        for (int j = 0; j < 2; ++j)
            #pragma unroll
            for (int e = 0; e < 8; ++e) acc[i][j][e] = 0.f;

    // async-stage one BKx(BM,BN) slab into LDS buffer `buf`
    auto stage = [&](int buf, long long ko) {
        #pragma unroll
        for (int i = 0; i < 4; ++i)
            async_load_b128(aL[buf] + (unsigned)(32 * i * LDSA_STRIDE * 2),
                            aG + (long long)(32 * i) * p.lda + ko);
        #pragma unroll
        for (int i = 0; i < 2; ++i)
            async_load_b128(bL[buf] + (unsigned)(32 * i * LDSB_STRIDE * 2),
                            bG + (long long)(32 * i) * p.ldb + ko);
    };

    // ---- prologue: stage 0 ----
    stage(0, 0);
    wait_async0();
    __syncthreads();

    const int nstages = p.K / BK;
    const int lr = lane & 15;
    const int ka = (lane >> 4) * 8;      // A frag: K 0..7/16..23 (lo lanes), 8..15/24..31 (hi)
    const int kb = (lane >> 4) * 8;      // B frag: K 0..7/16..23 (lo lanes), 8..15/24..31 (hi)

    for (int s = 0; s < nstages; ++s) {
        const int cur = s & 1, nxt = cur ^ 1;
        const bool pre = (s + 1 < nstages);
        // issue next-stage async loads early; latency hidden behind 8 WMMAs
        if (pre) stage(nxt, (long long)(s + 1) * BK);

        #pragma unroll
        for (int kk = 0; kk < BK; kk += 32) {
            union Frag { uint4 u[2]; v16bf v; } fa[2], fb[2];
            #pragma unroll
            for (int i = 0; i < 2; ++i) {
                const bf16_t* pA = &sA[cur][(wm + i * 16 + lr) * LDSA_STRIDE + kk + ka];
                fa[i].u[0] = *(const uint4*)(pA);
                fa[i].u[1] = *(const uint4*)(pA + 16);
            }
            #pragma unroll
            for (int j = 0; j < 2; ++j) {
                const bf16_t* pB = &sB[cur][(wn + j * 16 + lr) * LDSB_STRIDE + kk + kb];
                fb[j].u[0] = *(const uint4*)(pB);
                fb[j].u[1] = *(const uint4*)(pB + 16);
            }
            #pragma unroll
            for (int i = 0; i < 2; ++i)
                #pragma unroll
                for (int j = 0; j < 2; ++j)
                    acc[i][j] = __builtin_amdgcn_wmma_f32_16x16x32_bf16(
                        false, fa[i].v, false, fb[j].v, (short)0, acc[i][j], false, false);
        }
        if (pre) wait_async0();
        __syncthreads();
    }

    // -- epilogue: alpha, bias, optional exact GELU, strided fp32/bf16 store --
    const long long cOff = (long long)blockIdx.z * p.strC;
    const int nl = lane & 15, ro = (lane >> 4) * 8;
    #pragma unroll
    for (int j = 0; j < 2; ++j) {
        long long n = n0 + wn + j * 16 + nl;
        float bv = p.bias ? p.bias[n] : 0.f;
        #pragma unroll
        for (int i = 0; i < 2; ++i) {
            #pragma unroll
            for (int r = 0; r < 8; ++r) {
                long long m = m0 + wm + i * 16 + ro + r;
                float val = p.alpha * acc[i][j][r] + bv;
                if (p.act_gelu) val = 0.5f * val * (1.0f + erff(val * 0.70710678f));
                long long off = cOff + m * p.ldcM + n * p.ldcN;
                if (p.out_bf16) ((bf16_t*)p.C)[off] = (bf16_t)val;
                else            ((float*)p.C)[off]  = val;
            }
        }
    }
}

// ---------------------------------------------------------------------------
// Host side
// ---------------------------------------------------------------------------
static inline size_t alignup(size_t x) { return (x + 255) & ~(size_t)255; }

static void gemm(hipStream_t stream, const bf16_t* A, const bf16_t* B, void* C,
                 const float* bias, int M, int N, int K,
                 long long lda, long long ldb, long long ldcM, long long ldcN,
                 long long strA, long long strB, long long strC, int batch,
                 float alpha, int out_bf16, int gelu) {
    GemmP p{A, B, C, bias, M, N, K, lda, ldb, ldcM, ldcN, strA, strB, strC,
            alpha, out_bf16, gelu};
    dim3 grid((N + BN - 1) / BN, (M + BM - 1) / BM, batch);
    k_gemm_wmma<<<grid, dim3(256), 0, stream>>>(p);
}

extern "C" void kernel_launch(void* const* d_in, const int* in_sizes, int n_in,
                              void* d_out, int out_size, void* d_ws, size_t ws_size,
                              hipStream_t stream) {
    (void)in_sizes; (void)n_in; (void)out_size; (void)ws_size;
    const int Bb = 4, S = 1024, Dm = 1024, Hh = 16, Ff = 4096, Ll = 4, DH = 64, P = 1024;
    const long long BS = (long long)Bb * S;
    const float scale = 1.0f / sqrtf((float)(DH * 3));   // 1/sqrt(192)

    const int*   ids   = (const int*)d_in[0];
    const int*   segs  = (const int*)d_in[1];
    const float* amask = (const float*)d_in[2];
    const float* tok   = (const float*)d_in[3];
    const float* segE  = (const float*)d_in[4];
    const float* elg   = (const float*)d_in[5];
    const float* elb   = (const float*)d_in[6];
    const float* rel   = (const float*)d_in[7];
    const float* Wq    = (const float*)d_in[8];
    const float* bq    = (const float*)d_in[9];
    const float* Wk    = (const float*)d_in[10];
    const float* bk    = (const float*)d_in[11];
    const float* Wv    = (const float*)d_in[12];
    const float* bv    = (const float*)d_in[13];
    const float* Wo    = (const float*)d_in[14];
    const float* bo    = (const float*)d_in[15];
    const float* l1g   = (const float*)d_in[16];
    const float* l1b   = (const float*)d_in[17];
    const float* W1    = (const float*)d_in[18];
    const float* b1    = (const float*)d_in[19];
    const float* W2    = (const float*)d_in[20];
    const float* b2    = (const float*)d_in[21];
    const float* l2g   = (const float*)d_in[22];
    const float* l2b   = (const float*)d_in[23];

    char* ws = (char*)d_ws;
    size_t off = 0;
    auto alloc = [&](size_t bytes) { void* pp = ws + off; off = alignup(off + bytes); return pp; };

    bf16_t* wqt  = (bf16_t*)alloc((size_t)Ll * Dm * Dm * 2);
    bf16_t* wkt  = (bf16_t*)alloc((size_t)Ll * Dm * Dm * 2);
    bf16_t* wvt  = (bf16_t*)alloc((size_t)Ll * Dm * Dm * 2);
    bf16_t* wot  = (bf16_t*)alloc((size_t)Ll * Dm * Dm * 2);
    bf16_t* w1t  = (bf16_t*)alloc((size_t)Ll * Dm * Ff * 2);   // [F, D]
    bf16_t* w2t  = (bf16_t*)alloc((size_t)Ll * Ff * Dm * 2);   // [D, F]
    bf16_t* relb = (bf16_t*)alloc((size_t)P * Dm * 2);
    float*  hf   = (float*) alloc((size_t)BS * Dm * 4);
    bf16_t* hb   = (bf16_t*)alloc((size_t)BS * Dm * 2);
    bf16_t* qb   = (bf16_t*)alloc((size_t)BS * Dm * 2);
    bf16_t* kb2  = (bf16_t*)alloc((size_t)BS * Dm * 2);
    bf16_t* vtb  = (bf16_t*)alloc((size_t)BS * Dm * 2);        // [B,H,DH,S]
    bf16_t* pkb  = (bf16_t*)alloc((size_t)P * Dm * 2);
    bf16_t* pqb  = (bf16_t*)alloc((size_t)P * Dm * 2);
    float*  qkf  = (float*) alloc((size_t)Hh * S * S * 4);
    float*  c2pf = (float*) alloc((size_t)Hh * S * P * 4);
    float*  p2cf = (float*) alloc((size_t)Hh * S * P * 4);
    bf16_t* prb  = (bf16_t*)alloc((size_t)Hh * S * S * 2);
    bf16_t* ctxb = (bf16_t*)alloc((size_t)BS * Dm * 2);
    float*  dlt  = (float*) alloc((size_t)BS * Dm * 4);        // attn_out / ff2 (reused)
    bf16_t* ff1b = (bf16_t*)alloc((size_t)BS * Ff * 2);

    // ---- one-time: weight transpose+cvt, rel table cvt, embedding ----
    auto T = [&](const float* src, bf16_t* dst, int R, int C) {
        k_transpose_cvt<<<dim3((C + 31) / 32, (R + 31) / 32), dim3(256), 0, stream>>>(src, dst, R, C);
    };
    for (int l = 0; l < Ll; ++l) {
        long long od = (long long)l * Dm * Dm, of = (long long)l * Dm * Ff;
        T(Wq + od, wqt + od, Dm, Dm);
        T(Wk + od, wkt + od, Dm, Dm);
        T(Wv + od, wvt + od, Dm, Dm);
        T(Wo + od, wot + od, Dm, Dm);
        T(W1 + of, w1t + of, Dm, Ff);     // [D,F] -> [F,D]
        T(W2 + of, w2t + of, Ff, Dm);     // [F,D] -> [D,F]
    }
    {
        long long n = (long long)P * Dm;
        k_cvt_bf16<<<dim3((unsigned)((n / 4 + 255) / 256)), dim3(256), 0, stream>>>(rel, relb, n);
    }
    k_embed_ln<<<dim3((unsigned)BS), dim3(WG), 0, stream>>>(
        ids, segs, amask, tok, segE, elg, elb, hf, hb, Dm);

    // ---- layers ----
    for (int l = 0; l < Ll; ++l) {
        const bf16_t* wq_l = wqt + (long long)l * Dm * Dm;
        const bf16_t* wk_l = wkt + (long long)l * Dm * Dm;
        const bf16_t* wv_l = wvt + (long long)l * Dm * Dm;
        const bf16_t* wo_l = wot + (long long)l * Dm * Dm;
        const bf16_t* w1_l = w1t + (long long)l * Dm * Ff;
        const bf16_t* w2_l = w2t + (long long)l * Dm * Ff;
        const float* bq_l = bq + (long long)l * Dm;
        const float* bk_l = bk + (long long)l * Dm;
        const float* bv_l = bv + (long long)l * Dm;
        const float* bo_l = bo + (long long)l * Dm;
        const float* b1_l = b1 + (long long)l * Ff;
        const float* b2_l = b2 + (long long)l * Dm;

        // q, k projections -> bf16 [B*S, D]
        gemm(stream, hb, wq_l, qb,  bq_l, (int)BS, Dm, Dm, Dm, Dm, Dm, 1, 0, 0, 0, 1, 1.f, 1, 0);
        gemm(stream, hb, wk_l, kb2, bk_l, (int)BS, Dm, Dm, Dm, Dm, Dm, 1, 0, 0, 0, 1, 1.f, 1, 0);
        // v projection -> transposed bf16 [B,H,DH,S] (batch over b; C: m-stride 1, n-stride S)
        gemm(stream, hb, wv_l, vtb, bv_l, S, Dm, Dm, Dm, Dm, 1, S,
             (long long)S * Dm, 0, (long long)Dm * S, Bb, 1.f, 1, 0);
        // shared-projection position tables -> bf16 [P, D]
        gemm(stream, relb, wk_l, pkb, bk_l, P, Dm, Dm, Dm, Dm, Dm, 1, 0, 0, 0, 1, 1.f, 1, 0);
        gemm(stream, relb, wq_l, pqb, bq_l, P, Dm, Dm, Dm, Dm, Dm, 1, 0, 0, 0, 1, 1.f, 1, 0);

        for (int b = 0; b < Bb; ++b) {
            const bf16_t* qbB = qb  + (long long)b * S * Dm;
            const bf16_t* kbB = kb2 + (long long)b * S * Dm;
            // qk[h,q,k] = scale * q . k   (batched over heads)
            gemm(stream, qbB, kbB, qkf, nullptr, S, S, DH, Dm, Dm, S, 1,
                 DH, DH, (long long)S * S, Hh, scale, 0, 0);
            // c2p[h,q,p] = scale * q . pos_k
            gemm(stream, qbB, pkb, c2pf, nullptr, S, P, DH, Dm, Dm, P, 1,
                 DH, DH, (long long)S * P, Hh, scale, 0, 0);
            // p2c[h,k,p] = scale * k . pos_q
            gemm(stream, kbB, pqb, p2cf, nullptr, S, P, DH, Dm, Dm, P, 1,
                 DH, DH, (long long)S * P, Hh, scale, 0, 0);
            // fused gathers + mask + softmax -> bf16 probs
            k_attn_softmax<<<dim3(S, Hh), dim3(WG), 0, stream>>>(
                qkf, c2pf, p2cf, amask + (long long)b * S, prb, S, P);
            // ctx[b,q,h*64+d] = probs @ v   (B operand is vT, NT form; batch over heads)
            gemm(stream, prb, vtb + (long long)b * Dm * S,
                 ctxb + (long long)b * S * Dm, nullptr, S, DH, S, S, S, Dm, 1,
                 (long long)S * S, (long long)DH * S, DH, Hh, 1.f, 1, 0);
        }

        // output projection -> fp32 delta
        gemm(stream, ctxb, wo_l, dlt, bo_l, (int)BS, Dm, Dm, Dm, Dm, Dm, 1, 0, 0, 0, 1, 1.f, 0, 0);
        // h = LN(h + delta)
        k_ln_residual<<<dim3((unsigned)BS), dim3(WG), 0, stream>>>(
            hf, dlt, l1g + (long long)l * Dm, l1b + (long long)l * Dm, hf, hb, Dm);

        // FFN: GELU(h @ W1 + b1) @ W2 + b2
        gemm(stream, hb, w1_l, ff1b, b1_l, (int)BS, Ff, Dm, Dm, Dm, Ff, 1, 0, 0, 0, 1, 1.f, 1, 1);
        gemm(stream, ff1b, w2_l, dlt, b2_l, (int)BS, Dm, Ff, Ff, Ff, Dm, 1, 0, 0, 0, 1, 1.f, 0, 0);
        float* dst = (l == Ll - 1) ? (float*)d_out : hf;
        k_ln_residual<<<dim3((unsigned)BS), dim3(WG), 0, stream>>>(
            hf, dlt, l2g + (long long)l * Dm, l2b + (long long)l * Dm, dst, hb, Dm);
    }
}
